// Net_SLSTM_88347477278799
// MI455X (gfx1250) — matrix operand order repaired and checked by reading, not compiled, not run on hardware
//
#include <hip/hip_runtime.h>
#include <hip/hip_bf16.h>
#include <math.h>

// ---------------------------------------------------------------------------
// SLSTM (snntorch) 2-layer forward for MI455X (gfx1250, wave32, WMMA).
//  T=1000 B=256 C=14 H=128 NC=7.  Sequential recurrence -> 1 kernel per step,
//  each fusing  gates = x_t@Wih^T + mem@Whh^T + bias  (bf16 WMMA, f32 acc)
//  with the LSTM/spike epilogue in registers.  The epilogue also emits a bf16
//  shadow of mem so the next step's WMMA A-operand loads need no conversions.
// ---------------------------------------------------------------------------

#define Tn   1000
#define Bsz  256
#define Cn   14
#define Hn   128
#define NCn  7
#define G4H  512
#define BN_EPS 1e-5f

typedef __attribute__((ext_vector_type(16))) __bf16 v16bf;
typedef __attribute__((ext_vector_type(8)))  __bf16 v8bf;
typedef __attribute__((ext_vector_type(8)))  float  v8f;

__device__ __forceinline__ float sigmf(float x) { return 1.0f / (1.0f + __expf(-x)); }
__device__ __forceinline__ __bf16 f2bf(float f) { return (__bf16)f; }

// ---------------- fragment layouts (per CDNA5 ISA 7.12.2, wave32) -----------
// A (16x32 bf16): lane<16: row M=lane, e0..7 -> K=e, e8..15 -> K=16+(e-8)
//                 lane>=16: row M=lane-16, e0..7 -> K=8+e, e8..15 -> K=24+(e-8)
// B (32x16 bf16): col N=lane&15, e -> K=(lane>=16?8:0)+(e&7)+(e>=8?16:0)
// C/D (16x16 f32): element j: M=j+8*(lane>=16), N=lane&15
// Packed weight buffers: frag[ntile][kt][lane][16] bf16 (32B per lane-chunk).

// per-lane A fragment from a bf16 row-major [.,128] matrix
__device__ __forceinline__ v16bf loadA_bf16(const __bf16* rowBase /* + k0 */)
{
    const v8bf lo = *(const v8bf*)(rowBase);
    const v8bf hi = *(const v8bf*)(rowBase + 16);
    return __builtin_shufflevector(lo, hi, 0, 1, 2, 3, 4, 5, 6, 7,
                                           8, 9, 10, 11, 12, 13, 14, 15);
}

// ---------------------------------------------------------------------------
__global__ void k_prep(const float* __restrict__ Wih1, const float* __restrict__ Whh1,
                       const float* __restrict__ bih1, const float* __restrict__ bhh1,
                       const float* __restrict__ Whh2,
                       const float* __restrict__ bih2, const float* __restrict__ bhh2,
                       __bf16* __restrict__ fragRec1, __bf16* __restrict__ fragRec2,
                       __bf16* __restrict__ fragIn1,
                       float* __restrict__ bias1, float* __restrict__ bias2,
                       float* __restrict__ stateBase, __bf16* __restrict__ memBfBase,
                       float* __restrict__ bnsum)
{
    const int gid = blockIdx.x * blockDim.x + threadIdx.x;
    const int nt  = gridDim.x * blockDim.x;

    // zero syn1, mem1a, mem1b, syn2, mem2a, mem2b, memacc (contiguous f32)
    for (int i = gid; i < 7 * Bsz * Hn; i += nt) stateBase[i] = 0.0f;
    // zero bf16 shadows: mem1bfA, mem1bfB, mem2bfA, mem2bfB (contiguous)
    for (int i = gid; i < 4 * Bsz * Hn; i += nt) memBfBase[i] = f2bf(0.0f);
    for (int i = gid; i < Hn; i += nt) bnsum[i] = 0.0f;
    for (int i = gid; i < G4H; i += nt) {
        bias1[i] = bih1[i] + bhh1[i];
        bias2[i] = bih2[i] + bhh2[i];
    }
    // pack recurrent weights (32 ntiles x 4 ktiles x 32 lanes x 16 elems)
    for (int i = gid; i < 32 * 4 * 32 * 16; i += nt) {
        const int e     = i & 15;
        const int lane  = (i >> 4) & 31;
        const int kt    = (i >> 9) & 3;
        const int ntile = i >> 11;
        const int n = ntile * 16 + (lane & 15);
        const int k = kt * 32 + ((lane >= 16) ? 8 : 0) + (e & 7) + ((e >= 8) ? 16 : 0);
        fragRec1[i] = f2bf(Whh1[n * Hn + k]);
        fragRec2[i] = f2bf(Whh2[n * Hn + k]);
    }
    // pack layer-1 input weights, K padded 14 -> 32 (one ktile)
    for (int i = gid; i < 32 * 32 * 16; i += nt) {
        const int e     = i & 15;
        const int lane  = (i >> 4) & 31;
        const int ntile = i >> 9;
        const int n = ntile * 16 + (lane & 15);
        const int k = ((lane >= 16) ? 8 : 0) + (e & 7) + ((e >= 8) ? 16 : 0);
        fragIn1[i] = (k < Cn) ? f2bf(Wih1[n * Cn + k]) : f2bf(0.0f);
    }
}

// ---------------------------------------------------------------------------
// Layer-1 step: gates = bias1 + x_t@Wih1^T + mem@Whh1^T  -> LSTM update, spike
__global__ void __launch_bounds__(256)
k_step1(int t, const float* __restrict__ x,
        const __bf16* __restrict__ fragIn1, const __bf16* __restrict__ fragRec1,
        const float* __restrict__ bias1, const float* __restrict__ thrp,
        float* __restrict__ syn,
        const float* __restrict__ memIn, const __bf16* __restrict__ memInBf,
        float* __restrict__ memOut, __bf16* __restrict__ memOutBf,
        unsigned char* __restrict__ spkOut)
{
    const int lane  = threadIdx.x & 31;
    const int wid   = threadIdx.x >> 5;
    const int w     = blockIdx.x * 8 + wid;   // 128 waves total
    const int mtile = w >> 3;                 // 0..15 (batch tiles)
    const int htile = w & 7;                  // 0..7  (hidden tiles)
    const int hi    = lane >> 4;
    const int lm    = lane & 15;

    v8f acc[4];
#pragma unroll
    for (int g = 0; g < 4; ++g) {
        const float bv = bias1[g * Hn + htile * 16 + lm];
#pragma unroll
        for (int j = 0; j < 8; ++j) acc[g][j] = bv;
    }

    const int brow = mtile * 16 + lm;

    // recurrent GEMM: K = 128, A = bf16 shadow of mem (no conversions)
#pragma unroll
    for (int kt = 0; kt < 4; ++kt) {
        const v16bf A = loadA_bf16(memInBf + (size_t)brow * Hn + kt * 32 + hi * 8);
#pragma unroll
        for (int g = 0; g < 4; ++g) {
            const int ntile = g * 8 + htile;
            const v16bf Bf = *(const v16bf*)(fragRec1 + ((size_t)(ntile * 4 + kt) * 32 + lane) * 16);
            acc[g] = __builtin_amdgcn_wmma_f32_16x16x32_bf16(false, A, false, Bf,
                                                             (short)0, acc[g], false, false);
        }
    }

    // input GEMM: one ktile, K padded 14 -> 32
    {
        const float* px = x + ((size_t)t * Bsz + brow) * Cn;
        const int kb = hi * 8;
        v16bf A;
#pragma unroll
        for (int e = 0; e < 8; ++e) {
            const int k = kb + e;
            A[e]     = (k < Cn) ? f2bf(px[k]) : f2bf(0.0f);
            A[e + 8] = f2bf(0.0f);   // K >= 16 region is zero padding
        }
#pragma unroll
        for (int g = 0; g < 4; ++g) {
            const int ntile = g * 8 + htile;
            const v16bf Bf = *(const v16bf*)(fragIn1 + ((size_t)ntile * 32 + lane) * 16);
            acc[g] = __builtin_amdgcn_wmma_f32_16x16x32_bf16(false, A, false, Bf,
                                                             (short)0, acc[g], false, false);
        }
    }

    // epilogue (C layout: M = j + 8*hi, N = lm)
    const float thr = thrp[0];
#pragma unroll
    for (int j = 0; j < 8; ++j) {
        const int b = mtile * 16 + j + hi * 8;
        const int h = htile * 16 + lm;
        const size_t idx = (size_t)b * Hn + h;
        const float iG = acc[0][j], fG = acc[1][j], gG = acc[2][j], oG = acc[3][j];
        const float sv    = syn[idx];
        const float mprev = memIn[idx];
        const float reset = (mprev > thr) ? thr : 0.0f;     // detached reset*thr
        const float c  = sigmf(fG) * sv + sigmf(iG) * tanhf(gG);
        const float hn = sigmf(oG) * tanhf(c) - reset;
        syn[idx]      = c;
        memOut[idx]   = hn;
        memOutBf[idx] = f2bf(hn);                 // bf16 shadow for next GEMM
        spkOut[(size_t)t * Bsz * Hn + idx] = (hn > thr) ? 1u : 0u;
    }
}

// ---------------------------------------------------------------------------
// BN statistics over binary spikes: per-channel sum (mean=p, var=p-p^2)
__global__ void k_bnsum(const unsigned char* __restrict__ spk, float* __restrict__ bnsum)
{
    __shared__ float s[256];
    const int tid  = threadIdx.x;
    const int h    = tid & 127;
    const int half = tid >> 7;
    float acc = 0.0f;
    for (long long r = (long long)blockIdx.x * 2 + half; r < (long long)Tn * Bsz;
         r += (long long)gridDim.x * 2)
        acc += (float)spk[(size_t)r * Hn + h];
    s[tid] = acc;
    __syncthreads();
    if (tid < 128) atomicAdd(&bnsum[tid], s[tid] + s[tid + 128]);
}

// scale s_h = gamma*rsqrt(var+eps); shift t_h = beta - p*s_h; fold shift into
// the layer-2 gate bias: tb2[n] = bias2[n] + sum_k t_k * Wih2[n,k]
__global__ void k_bnfinal(const float* __restrict__ bnsum,
                          const float* __restrict__ gamma, const float* __restrict__ beta,
                          const float* __restrict__ Wih2, const float* __restrict__ bias2,
                          float* __restrict__ sArr, float* __restrict__ tb2)
{
    __shared__ float tSh[128];
    const int tid = threadIdx.x;   // 512 threads
    if (tid < 128) {
        const float p   = bnsum[tid] * (1.0f / ((float)Tn * (float)Bsz));
        const float var = p - p * p;                 // biased var of Bernoulli
        const float sc  = gamma[tid] * rsqrtf(var + BN_EPS);
        const float tv  = beta[tid] - p * sc;
        sArr[tid] = sc;
        tSh[tid]  = tv;
    }
    __syncthreads();
    float sum = bias2[tid];
    for (int k = 0; k < Hn; ++k) sum += tSh[k] * Wih2[tid * Hn + k];
    tb2[tid] = sum;
}

// pack BN-scaled layer-2 input weights: W2s[k,n] = s_k * Wih2[n,k]
__global__ void k_pack2(const float* __restrict__ Wih2, const float* __restrict__ sArr,
                        __bf16* __restrict__ fragIn2)
{
    const int gid = blockIdx.x * blockDim.x + threadIdx.x;
    const int nt  = gridDim.x * blockDim.x;
    for (int i = gid; i < 32 * 4 * 32 * 16; i += nt) {
        const int e     = i & 15;
        const int lane  = (i >> 4) & 31;
        const int kt    = (i >> 9) & 3;
        const int ntile = i >> 11;
        const int n = ntile * 16 + (lane & 15);
        const int k = kt * 32 + ((lane >= 16) ? 8 : 0) + (e & 7) + ((e >= 8) ? 16 : 0);
        fragIn2[i] = f2bf(sArr[k] * Wih2[n * Hn + k]);
    }
}

// ---------------------------------------------------------------------------
// Layer-2 step: gates = tb2 + spk_t@W2s + mem@Whh2^T ; accumulate h into memacc
__global__ void __launch_bounds__(256)
k_step2(int t, const unsigned char* __restrict__ spk,
        const __bf16* __restrict__ fragIn2, const __bf16* __restrict__ fragRec2,
        const float* __restrict__ tb2, const float* __restrict__ thrp,
        float* __restrict__ syn,
        const float* __restrict__ memIn, const __bf16* __restrict__ memInBf,
        float* __restrict__ memOut, __bf16* __restrict__ memOutBf,
        float* __restrict__ memacc)
{
    const int lane  = threadIdx.x & 31;
    const int wid   = threadIdx.x >> 5;
    const int w     = blockIdx.x * 8 + wid;
    const int mtile = w >> 3;
    const int htile = w & 7;
    const int hi    = lane >> 4;
    const int lm    = lane & 15;

    v8f acc[4];
#pragma unroll
    for (int g = 0; g < 4; ++g) {
        const float bv = tb2[g * Hn + htile * 16 + lm];
#pragma unroll
        for (int j = 0; j < 8; ++j) acc[g][j] = bv;
    }

    const int brow = mtile * 16 + lm;

    // input GEMM: K = 128, A = binary spikes (u8 -> bf16, exact; VALU work
    // co-executes with the XDL WMMA pipe)
#pragma unroll
    for (int kt = 0; kt < 4; ++kt) {
        const unsigned char* ps = spk + ((size_t)t * Bsz + brow) * Hn + kt * 32 + hi * 8;
        const unsigned long long r0 = *(const unsigned long long*)(ps);
        const unsigned long long r1 = *(const unsigned long long*)(ps + 16);
        v16bf A;
#pragma unroll
        for (int e = 0; e < 8; ++e) {
            A[e]     = f2bf((float)((r0 >> (8 * e)) & 0xffull));
            A[e + 8] = f2bf((float)((r1 >> (8 * e)) & 0xffull));
        }
#pragma unroll
        for (int g = 0; g < 4; ++g) {
            const int ntile = g * 8 + htile;
            const v16bf Bf = *(const v16bf*)(fragIn2 + ((size_t)(ntile * 4 + kt) * 32 + lane) * 16);
            acc[g] = __builtin_amdgcn_wmma_f32_16x16x32_bf16(false, A, false, Bf,
                                                             (short)0, acc[g], false, false);
        }
    }

    // recurrent GEMM: K = 128, A = bf16 shadow of mem
#pragma unroll
    for (int kt = 0; kt < 4; ++kt) {
        const v16bf A = loadA_bf16(memInBf + (size_t)brow * Hn + kt * 32 + hi * 8);
#pragma unroll
        for (int g = 0; g < 4; ++g) {
            const int ntile = g * 8 + htile;
            const v16bf Bf = *(const v16bf*)(fragRec2 + ((size_t)(ntile * 4 + kt) * 32 + lane) * 16);
            acc[g] = __builtin_amdgcn_wmma_f32_16x16x32_bf16(false, A, false, Bf,
                                                             (short)0, acc[g], false, false);
        }
    }

    const float thr = thrp[0];
#pragma unroll
    for (int j = 0; j < 8; ++j) {
        const int b = mtile * 16 + j + hi * 8;
        const int h = htile * 16 + lm;
        const size_t idx = (size_t)b * Hn + h;
        const float iG = acc[0][j], fG = acc[1][j], gG = acc[2][j], oG = acc[3][j];
        const float sv    = syn[idx];
        const float mprev = memIn[idx];
        const float reset = (mprev > thr) ? thr : 0.0f;
        const float c  = sigmf(fG) * sv + sigmf(iG) * tanhf(gG);
        const float hn = sigmf(oG) * tanhf(c) - reset;
        syn[idx]      = c;
        memOut[idx]   = hn;
        memOutBf[idx] = f2bf(hn);
        memacc[idx]  += hn;     // running sum over T for finalMem
    }
}

// ---------------------------------------------------------------------------
// head: out[b,nc] = (memacc[b,:]/T) . Wfc[nc,:] + bfc[nc]
__global__ void k_head(const float* __restrict__ memacc, const float* __restrict__ Wfc,
                       const float* __restrict__ bfc, float* __restrict__ out)
{
    __shared__ float s[128];
    const int b = blockIdx.x;
    const int h = threadIdx.x;   // 128 threads
    const float m = memacc[(size_t)b * Hn + h] * (1.0f / (float)Tn);
    for (int nc = 0; nc < NCn; ++nc) {
        s[h] = m * Wfc[nc * Hn + h];
        __syncthreads();
        for (int off = 64; off > 0; off >>= 1) {
            if (h < off) s[h] += s[h + off];
            __syncthreads();
        }
        if (h == 0) out[b * NCn + nc] = s[0] + bfc[nc];
        __syncthreads();
    }
}

// ---------------------------------------------------------------------------
extern "C" void kernel_launch(void* const* d_in, const int* in_sizes, int n_in,
                              void* d_out, int out_size, void* d_ws, size_t ws_size,
                              hipStream_t stream)
{
    (void)in_sizes; (void)n_in; (void)out_size; (void)ws_size;

    const float* x     = (const float*)d_in[0];
    const float* Wih1  = (const float*)d_in[1];
    const float* Whh1  = (const float*)d_in[2];
    const float* bih1  = (const float*)d_in[3];
    const float* bhh1  = (const float*)d_in[4];
    const float* thr1  = (const float*)d_in[5];
    const float* Wih2  = (const float*)d_in[6];
    const float* Whh2  = (const float*)d_in[7];
    const float* bih2  = (const float*)d_in[8];
    const float* bhh2  = (const float*)d_in[9];
    const float* thr2  = (const float*)d_in[10];
    const float* gamma = (const float*)d_in[11];
    const float* beta  = (const float*)d_in[12];
    const float* Wfc   = (const float*)d_in[13];
    const float* bfc   = (const float*)d_in[14];
    float* out = (float*)d_out;

    // ---- workspace carve-up (all offsets 256B aligned) ----
    char* p = (char*)d_ws;
    __bf16* fragRec1  = (__bf16*)(p + 0);          // 131072 B
    __bf16* fragRec2  = (__bf16*)(p + 131072);     // 131072 B
    __bf16* fragIn1   = (__bf16*)(p + 262144);     //  32768 B
    __bf16* fragIn2   = (__bf16*)(p + 294912);     // 131072 B
    float*  bias1     = (float*)(p + 425984);      //   2048 B
    float*  bias2     = (float*)(p + 428032);      //   2048 B
    float*  tb2       = (float*)(p + 430080);      //   2048 B
    float*  sArr      = (float*)(p + 432128);      //    512 B
    float*  bnsum     = (float*)(p + 432640);      //    512 B
    float*  stateBase = (float*)(p + 433152);      // 7 * 131072 B
    float*  syn1      = stateBase;
    float*  mem1a     = stateBase + 32768;
    float*  mem1b     = stateBase + 65536;
    float*  syn2      = stateBase + 98304;
    float*  mem2a     = stateBase + 131072;
    float*  mem2b     = stateBase + 163840;
    float*  memacc    = stateBase + 196608;
    __bf16* memBfBase = (__bf16*)(p + 1350656);    // 4 * 65536 B bf16 shadows
    __bf16* mem1bfA   = memBfBase;
    __bf16* mem1bfB   = memBfBase + 32768;
    __bf16* mem2bfA   = memBfBase + 65536;
    __bf16* mem2bfB   = memBfBase + 98304;
    unsigned char* spk = (unsigned char*)(p + 1612800); // 32,768,000 B

    // one-time: pack weights (bf16 fragment layout), fold biases, zero state
    k_prep<<<256, 256, 0, stream>>>(Wih1, Whh1, bih1, bhh1, Whh2, bih2, bhh2,
                                    fragRec1, fragRec2, fragIn1,
                                    bias1, bias2, stateBase, memBfBase, bnsum);

    // layer 1: sequential recurrence, mem ping-pong (f32 + bf16 shadow)
    for (int t = 0; t < Tn; ++t) {
        const float*  mi  = (t & 1) ? mem1b  : mem1a;
        float*        mo  = (t & 1) ? mem1a  : mem1b;
        const __bf16* mib = (t & 1) ? mem1bfB : mem1bfA;
        __bf16*       mob = (t & 1) ? mem1bfA : mem1bfB;
        k_step1<<<16, 256, 0, stream>>>(t, x, fragIn1, fragRec1, bias1, thr1,
                                        syn1, mi, mib, mo, mob, spk);
    }

    // BatchNorm stats over binary spikes + fold into layer-2 GEMM operands
    k_bnsum<<<128, 256, 0, stream>>>(spk, bnsum);
    k_bnfinal<<<1, 512, 0, stream>>>(bnsum, gamma, beta, Wih2, bias2, sArr, tb2);
    k_pack2<<<128, 256, 0, stream>>>(Wih2, sArr, fragIn2);

    // layer 2: sequential recurrence, accumulate h into memacc
    for (int t = 0; t < Tn; ++t) {
        const float*  mi  = (t & 1) ? mem2b  : mem2a;
        float*        mo  = (t & 1) ? mem2a  : mem2b;
        const __bf16* mib = (t & 1) ? mem2bfB : mem2bfA;
        __bf16*       mob = (t & 1) ? mem2bfA : mem2bfB;
        k_step2<<<16, 256, 0, stream>>>(t, spk, fragIn2, fragRec2, tb2, thr2,
                                        syn2, mi, mib, mo, mob, memacc);
    }

    // FC head
    k_head<<<Bsz, 128, 0, stream>>>(memacc, Wfc, bfc, out);
}
